// ActiveDendriteLayer_57904749084764
// MI455X (gfx1250) — matrix operand order repaired and checked by reading, not compiled, not run on hardware
//
#include <hip/hip_runtime.h>

#define B_SZ     4096
#define D_SZ     2048
#define NCELL    4096
#define NDEND    3
#define TDEND    (NCELL * NDEND)   // 12288

#define RG  4            // row-groups (of 16) per wave -> 64 rows/wave
#define BM  (8 * RG * 16) // 512 rows per block (8 waves)
#define BN  48           // dendrite columns per block (= 16 cells)
#define BK  32           // K step (bf16 WMMA K)

typedef __attribute__((ext_vector_type(16))) __bf16 v16bf;
typedef __attribute__((ext_vector_type(8)))  __bf16 v8bf;
typedef __attribute__((ext_vector_type(8)))  float  v8f;

// ---------------------------------------------------------------------------
// f32 -> bf16 conversion pass (one time; output lives in d_ws, L2-resident)
// ---------------------------------------------------------------------------
__global__ __launch_bounds__(256) void cvt_f32_bf16_kernel(
    const float* __restrict__ src, __bf16* __restrict__ dst, int n8)
{
    const int i = blockIdx.x * blockDim.x + threadIdx.x;
    if (i >= n8) return;
    const float4* s = (const float4*)src + (size_t)i * 2;
    const float4 u = s[0];
    const float4 v = s[1];
    v8bf o;
    o[0] = (__bf16)u.x; o[1] = (__bf16)u.y; o[2] = (__bf16)u.z; o[3] = (__bf16)u.w;
    o[4] = (__bf16)v.x; o[5] = (__bf16)v.y; o[6] = (__bf16)v.z; o[7] = (__bf16)v.w;
    ((v8bf*)dst)[i] = o;
}

// ---------------------------------------------------------------------------
// bf16 fragment loads in WMMA register layout
// ---------------------------------------------------------------------------
// A operand: elems 0..7 = K run at base, elems 8..15 = K run at base+8
__device__ __forceinline__ v16bf loadA_bf16(const __bf16* __restrict__ base) {
    const v8bf* p = (const v8bf*)base;
    const v8bf lo = p[0];
    const v8bf hi = p[1];   // +8 elements
    v16bf a;
    #pragma unroll
    for (int i = 0; i < 8; ++i) { a[i] = lo[i]; a[i + 8] = hi[i]; }
    return a;
}
// B operand: 16 contiguous K values per lane
__device__ __forceinline__ v16bf loadB_bf16(const __bf16* __restrict__ base) {
    return *(const v16bf*)base;
}

// ---------------------------------------------------------------------------
// Main GEMM (bf16 operands) + fused bias/ReLU/per-cell reduction
// Each wave: 64 rows x 48 cols = 12 WMMA accumulators, software-pipelined,
// K-loop unrolled x2 so the ping-pong rotation becomes register renaming.
// ---------------------------------------------------------------------------
__global__ __launch_bounds__(256) void active_dendrite_bf16_kernel(
    const __bf16* __restrict__ xb,     // [B, D] bf16
    const __bf16* __restrict__ Wb,     // [TDEND, D] bf16
    const float*  __restrict__ bd,     // [TDEND]
    const float*  __restrict__ Wl,     // [NCELL, 3]
    const float*  __restrict__ bl,     // [NCELL]
    float* __restrict__ out)           // [B, NCELL]
{
    __shared__ float hbuf[8][32 * BN];   // per-wave 32-row tile, 2 phases (48 KB)

    const int lane  = threadIdx.x & 31;
    const int wave  = threadIdx.x >> 5;
    const int lhalf = lane >> 4;
    const int lmod  = lane & 15;

    const int m0 = blockIdx.x * BM + wave * (RG * 16);  // wave's 64-row base
    const int n0 = blockIdx.y * BN;                     // dendrite column base

    const int ashift = lhalf * 16;
    const int bshift = lhalf * 16;

    const __bf16* __restrict__ pa[RG];
    const __bf16* __restrict__ pb[3];
    #pragma unroll
    for (int i = 0; i < RG; ++i)
        pa[i] = xb + (size_t)(m0 + i * 16 + lmod) * D_SZ + ashift;
    #pragma unroll
    for (int j = 0; j < 3; ++j)
        pb[j] = Wb + (size_t)(n0 + j * 16 + lmod) * D_SZ + bshift;

    v8f acc[RG][3];
    #pragma unroll
    for (int i = 0; i < RG; ++i)
        #pragma unroll
        for (int j = 0; j < 3; ++j)
            acc[i][j] = (v8f){0.f,0.f,0.f,0.f,0.f,0.f,0.f,0.f};

    v16bf af[RG], bfr[3];
    #pragma unroll
    for (int i = 0; i < RG; ++i) af[i] = loadA_bf16(pa[i]);
    #pragma unroll
    for (int j = 0; j < 3; ++j) bfr[j] = loadB_bf16(pb[j]);

    #pragma unroll 2
    for (int k0 = 0; k0 < D_SZ; k0 += BK) {
        // software pipeline: issue next-iteration fragment loads first
        const int kn = (k0 + BK < D_SZ) ? (k0 + BK) : 0;
        v16bf na[RG], nb[3];
        #pragma unroll
        for (int i = 0; i < RG; ++i) na[i] = loadA_bf16(pa[i] + kn);
        #pragma unroll
        for (int j = 0; j < 3; ++j) nb[j] = loadB_bf16(pb[j] + kn);

        #pragma unroll
        for (int i = 0; i < RG; ++i)
            #pragma unroll
            for (int j = 0; j < 3; ++j)
                acc[i][j] = __builtin_amdgcn_wmma_f32_16x16x32_bf16(
                    false, af[i], false, bfr[j], (short)0, acc[i][j], false, false);

        #pragma unroll
        for (int i = 0; i < RG; ++i) af[i] = na[i];
        #pragma unroll
        for (int j = 0; j < 3; ++j) bfr[j] = nb[j];
    }

    // Epilogue: bias + ReLU -> LDS stash -> per-cell triple reduction.
    // Two 32-row phases to keep LDS at 48 KB.
    const float bdv[3] = { bd[n0 + lmod], bd[n0 + 16 + lmod], bd[n0 + 32 + lmod] };
    float* hb = hbuf[wave];
    const int rbase = lhalf * 8;
    const int cbase = blockIdx.y * 16;

    #pragma unroll
    for (int phase = 0; phase < 2; ++phase) {
        // stash row-groups {2*phase, 2*phase+1} -> local rows 0..31
        #pragma unroll
        for (int gg = 0; gg < 2; ++gg) {
            const int g = phase * 2 + gg;
            #pragma unroll
            for (int r = 0; r < 8; ++r) {
                const int row_l = gg * 16 + rbase + r;
                #pragma unroll
                for (int j = 0; j < 3; ++j) {
                    float h = acc[g][j][r] + bdv[j];
                    h = h > 0.f ? h : 0.f;
                    hb[row_l * BN + j * 16 + lmod] = h;
                }
            }
        }
        __syncthreads();

        // reduce 32 rows x 16 cells = 512 outputs, 16 per lane
        #pragma unroll
        for (int i = 0; i < 16; ++i) {
            const int idx   = i * 32 + lane;    // 0..511
            const int row_l = idx >> 4;         // 0..31
            const int cc    = idx & 15;         // 0..15
            const float* hp = hb + row_l * BN + cc * 3;
            const float h0 = hp[0], h1 = hp[1], h2 = hp[2];
            const int c = cbase + cc;
            const float o = h0 * Wl[c * 3 + 0] + h1 * Wl[c * 3 + 1] +
                            h2 * Wl[c * 3 + 2] + bl[c];
            out[(size_t)(m0 + phase * 32 + row_l) * NCELL + c] = o;
        }
        __syncthreads();
    }
}

// ---------------------------------------------------------------------------
// Fallback: fully fused f32->bf16-in-loop kernel (used if d_ws is too small)
// ---------------------------------------------------------------------------
__device__ __forceinline__ v16bf load_frag16_f32(const float* __restrict__ p) {
    float4 r0 = ((const float4*)p)[0];
    float4 r1 = ((const float4*)p)[1];
    float4 r2 = ((const float4*)p)[2];
    float4 r3 = ((const float4*)p)[3];
    v16bf v;
    v[0]  = (__bf16)r0.x; v[1]  = (__bf16)r0.y; v[2]  = (__bf16)r0.z; v[3]  = (__bf16)r0.w;
    v[4]  = (__bf16)r1.x; v[5]  = (__bf16)r1.y; v[6]  = (__bf16)r1.z; v[7]  = (__bf16)r1.w;
    v[8]  = (__bf16)r2.x; v[9]  = (__bf16)r2.y; v[10] = (__bf16)r2.z; v[11] = (__bf16)r2.w;
    v[12] = (__bf16)r3.x; v[13] = (__bf16)r3.y; v[14] = (__bf16)r3.z; v[15] = (__bf16)r3.w;
    return v;
}
__device__ __forceinline__ v16bf load_fragA_f32(const float* __restrict__ p,
                                                const float* __restrict__ q) {
    float4 r0 = ((const float4*)p)[0];
    float4 r1 = ((const float4*)p)[1];
    float4 r2 = ((const float4*)q)[0];
    float4 r3 = ((const float4*)q)[1];
    v16bf v;
    v[0]  = (__bf16)r0.x; v[1]  = (__bf16)r0.y; v[2]  = (__bf16)r0.z; v[3]  = (__bf16)r0.w;
    v[4]  = (__bf16)r1.x; v[5]  = (__bf16)r1.y; v[6]  = (__bf16)r1.z; v[7]  = (__bf16)r1.w;
    v[8]  = (__bf16)r2.x; v[9]  = (__bf16)r2.y; v[10] = (__bf16)r2.z; v[11] = (__bf16)r2.w;
    v[12] = (__bf16)r3.x; v[13] = (__bf16)r3.y; v[14] = (__bf16)r3.z; v[15] = (__bf16)r3.w;
    return v;
}

__global__ __launch_bounds__(256) void active_dendrite_fused_kernel(
    const float* __restrict__ x, const float* __restrict__ Wd,
    const float* __restrict__ bd, const float* __restrict__ Wl,
    const float* __restrict__ bl, float* __restrict__ out)
{
    __shared__ float hbuf[8][16 * BN];
    const int lane  = threadIdx.x & 31;
    const int wave  = threadIdx.x >> 5;
    const int lhalf = lane >> 4;
    const int lmod  = lane & 15;
    const int m0 = blockIdx.x * 128 + wave * 16;
    const int n0 = blockIdx.y * BN;
    const float* arow  = x  + (size_t)(m0 + lmod) * D_SZ;
    const float* brow0 = Wd + (size_t)(n0 +  0 + lmod) * D_SZ;
    const float* brow1 = Wd + (size_t)(n0 + 16 + lmod) * D_SZ;
    const float* brow2 = Wd + (size_t)(n0 + 32 + lmod) * D_SZ;
    v8f acc0 = {0.f,0.f,0.f,0.f,0.f,0.f,0.f,0.f};
    v8f acc1 = acc0, acc2 = acc0;
    const int ashift = lhalf * 16, bshift = lhalf * 16;
    for (int k0 = 0; k0 < D_SZ; k0 += BK) {
        __builtin_prefetch(arow + k0 + 2 * BK, 0, 0);
        v16bf a  = load_fragA_f32(arow + k0 + ashift, arow + k0 + 8 + ashift);
        v16bf b0 = load_frag16_f32(brow0 + k0 + bshift);
        v16bf b1 = load_frag16_f32(brow1 + k0 + bshift);
        v16bf b2 = load_frag16_f32(brow2 + k0 + bshift);
        acc0 = __builtin_amdgcn_wmma_f32_16x16x32_bf16(false, a, false, b0, (short)0, acc0, false, false);
        acc1 = __builtin_amdgcn_wmma_f32_16x16x32_bf16(false, a, false, b1, (short)0, acc1, false, false);
        acc2 = __builtin_amdgcn_wmma_f32_16x16x32_bf16(false, a, false, b2, (short)0, acc2, false, false);
    }
    const float bdv0 = bd[n0 + lmod], bdv1 = bd[n0 + 16 + lmod], bdv2 = bd[n0 + 32 + lmod];
    float* hb = hbuf[wave];
    const int rbase = lhalf * 8;
    #pragma unroll
    for (int r = 0; r < 8; ++r) {
        const int row = rbase + r;
        float h0 = acc0[r] + bdv0; h0 = h0 > 0.f ? h0 : 0.f;
        float h1 = acc1[r] + bdv1; h1 = h1 > 0.f ? h1 : 0.f;
        float h2 = acc2[r] + bdv2; h2 = h2 > 0.f ? h2 : 0.f;
        hb[row * BN + lmod] = h0; hb[row * BN + lmod + 16] = h1; hb[row * BN + lmod + 32] = h2;
    }
    __syncthreads();
    const int cbase = blockIdx.y * 16;
    #pragma unroll
    for (int i = 0; i < 8; ++i) {
        const int idx = i * 32 + lane;
        const int row = idx >> 4, cc = idx & 15;
        const float* hp = hb + row * BN + cc * 3;
        const int c = cbase + cc;
        out[(size_t)(m0 + row) * NCELL + c] =
            hp[0] * Wl[c*3+0] + hp[1] * Wl[c*3+1] + hp[2] * Wl[c*3+2] + bl[c];
    }
}

// ---------------------------------------------------------------------------
extern "C" void kernel_launch(void* const* d_in, const int* in_sizes, int n_in,
                              void* d_out, int out_size, void* d_ws, size_t ws_size,
                              hipStream_t stream) {
    const float* x  = (const float*)d_in[0];
    const float* Wd = (const float*)d_in[1];
    const float* bd = (const float*)d_in[2];
    const float* Wl = (const float*)d_in[3];
    const float* bl = (const float*)d_in[4];
    float* out = (float*)d_out;

    const size_t nx = (size_t)B_SZ * D_SZ;          //  8,388,608
    const size_t nw = (size_t)TDEND * D_SZ;         // 25,165,824
    const size_t need = (nx + nw) * sizeof(__bf16); // 64 MB

    if (ws_size >= need) {
        __bf16* xb = (__bf16*)d_ws;
        __bf16* Wb = xb + nx;
        {
            const int n8 = (int)(nx / 8);
            cvt_f32_bf16_kernel<<<(n8 + 255) / 256, 256, 0, stream>>>(x, xb, n8);
        }
        {
            const int n8 = (int)(nw / 8);
            cvt_f32_bf16_kernel<<<(n8 + 255) / 256, 256, 0, stream>>>(Wd, Wb, n8);
        }
        dim3 grid(B_SZ / BM, TDEND / BN);   // 8 x 256
        active_dendrite_bf16_kernel<<<grid, 256, 0, stream>>>(xb, Wb, bd, Wl, bl, out);
    } else {
        dim3 grid(B_SZ / 128, TDEND / BN);  // 32 x 256
        active_dendrite_fused_kernel<<<grid, 256, 0, stream>>>(x, Wd, bd, Wl, bl, out);
    }
}
